// SlidingWindowAttention_30846455120373
// MI455X (gfx1250) — compile-verified, matrix operand order checked by reading, not run on hardware
//
#include <hip/hip_runtime.h>

// ---------------------------------------------------------------------------
// Sliding-window attention (B=2, T=4096, C=1024, H=8, D=128, W=512) for
// MI455X / gfx1250.  All matmuls via v_wmma_f32_16x16x32_bf16 (fp32 accum).
// Flash-style attention (no T x T score matrix).  GEMMs use ping-pong
// register pipelining (no buffer-rotation copies); the output projection
// uses the gfx1250 async global->LDS copy path (ASYNCcnt).
// ---------------------------------------------------------------------------

#define B_   2
#define T_   4096
#define C_   1024
#define H_   8
#define D_   128
#define W_   512
#define TC_  3072               // 3*C
#define MT_  8192               // B*T

typedef __attribute__((ext_vector_type(16))) __bf16 v16bf;
typedef __attribute__((ext_vector_type(8)))  float  v8f;
typedef __attribute__((ext_vector_type(4)))  int    i4v;

#if __has_builtin(__builtin_amdgcn_global_load_async_to_lds_b128)
#define HAVE_ASYNC_LDS 1
#define AS1 __attribute__((address_space(1)))
#define AS3 __attribute__((address_space(3)))
// probe-revealed signature: (int4 AS1*, int4 AS3*, imm offset, imm cpol)
#define ASYNC_CP_B128(gsrc, ldst)                                            \
    __builtin_amdgcn_global_load_async_to_lds_b128(                          \
        (AS1 i4v*)(AS1 void*)(gsrc), (AS3 i4v*)(AS3 void*)(ldst), 0, 0)
#if __has_builtin(__builtin_amdgcn_s_wait_asynccnt)
#define WAIT_ASYNC(n) __builtin_amdgcn_s_wait_asynccnt(n)
#else
#define WAIT_ASYNC(n) asm volatile("s_wait_asynccnt %0" :: "i"(n) : "memory")
#endif
#endif

__device__ __forceinline__ unsigned short f32_to_bf16(float f) {
    unsigned int u = __float_as_uint(f);
    u += 0x7FFFu + ((u >> 16) & 1u);            // round-to-nearest-even
    return (unsigned short)(u >> 16);
}

__device__ __forceinline__ v8f wmma_bf16(v16bf a, v16bf b, v8f c) {
    // D = A(16x32 bf16) * B(32x16 bf16) + C(16x16 f32)
    return __builtin_amdgcn_wmma_f32_16x16x32_bf16(
        false, a, false, b, (short)0, c, false, false);
}

// ---------------------------------------------------------------------------
// Kernel 0: fp32 -> bf16 conversion, 4 elements/thread (packed b64 stores)
// ---------------------------------------------------------------------------
__global__ void cvt_f32_to_bf16_kernel(const float4* __restrict__ in,
                                       unsigned long long* __restrict__ out,
                                       int n4) {
    int i = blockIdx.x * blockDim.x + threadIdx.x;
    if (i < n4) {
        float4 f = in[i];
        unsigned long long p =
            (unsigned long long)f32_to_bf16(f.x)
          | ((unsigned long long)f32_to_bf16(f.y) << 16)
          | ((unsigned long long)f32_to_bf16(f.z) << 32)
          | ((unsigned long long)f32_to_bf16(f.w) << 48);
        out[i] = p;
    }
}

// ---------------------------------------------------------------------------
// Kernel 1: qkv = x @ w_qkv^T (bf16 WMMA) fused with RoPE.
// WG = 256 threads = 8 waves; tile 64(M) x 128(N); each N-tile is one
// (q|k|v, head) slab so the RoPE d<->d+64 pairing stays tile-local (LDS).
// K-loop unrolled x2 with ping-pong fragment sets: no rotation copies, and
// each stage's loads fly half an iteration ahead of their WMMAs.
// ---------------------------------------------------------------------------
__global__ __launch_bounds__(256) void qkv_rope_kernel(
    const unsigned short* __restrict__ xb,   // [8192,1024] bf16
    const unsigned short* __restrict__ wb,   // [3072,1024] bf16
    unsigned short* __restrict__ qo,         // [B,H,T,D]   bf16
    unsigned short* __restrict__ ko,         // [B,H,T,D]   bf16
    unsigned short* __restrict__ vo)         // [B,H,D,T]   bf16 (transposed)
{
    __shared__ float tile[64][132];          // +4 pad for pair reads

    const int tid   = threadIdx.x;
    const int wave  = tid >> 5;
    const int lane  = tid & 31;
    const int lrow  = lane & 15;
    const int khalf = lane >> 4;

    const int blk = blockIdx.x;
    const int bm  = blk & 127;               // 128 M-tiles of 64 rows
    const int bn  = blk >> 7;                // 24  N-tiles of 128 cols

    const int m0 = bm * 64  + (wave & 3) * 16;
    const int n0 = bn * 128 + (wave >> 2) * 64;

    v8f acc[4] = {};
    const unsigned short* arow = xb + (size_t)(m0 + lrow) * C_ + khalf * 16;
    const unsigned short* brow = wb + (size_t)(n0 + lrow) * C_ + khalf * 16;

    // ---- ping-pong pipelined K loop (2 stages of K=32 per iteration) ----
    v16bf a0 = *(const v16bf*)(arow);
    v16bf b0[4];
#pragma unroll
    for (int nt = 0; nt < 4; ++nt)
        b0[nt] = *(const v16bf*)(brow + (size_t)nt * 16 * C_);

    for (int k0 = 0; k0 < C_; k0 += 64) {
        __builtin_prefetch(arow + k0 + 256, 0, 3);
        // stage 1 loads (k0+32) in flight while stage-0 WMMAs run
        v16bf a1 = *(const v16bf*)(arow + k0 + 32);
        v16bf b1[4];
#pragma unroll
        for (int nt = 0; nt < 4; ++nt)
            b1[nt] = *(const v16bf*)(brow + (size_t)nt * 16 * C_ + k0 + 32);
#pragma unroll
        for (int nt = 0; nt < 4; ++nt)
            acc[nt] = wmma_bf16(a0, b0[nt], acc[nt]);
        // stage 0 loads for next iteration (k0+64) under stage-1 WMMAs
        if (k0 + 64 < C_) {
            a0 = *(const v16bf*)(arow + k0 + 64);
#pragma unroll
            for (int nt = 0; nt < 4; ++nt)
                b0[nt] = *(const v16bf*)(brow + (size_t)nt * 16 * C_ + k0 + 64);
        }
#pragma unroll
        for (int nt = 0; nt < 4; ++nt)
            acc[nt] = wmma_bf16(a1, b1[nt], acc[nt]);
    }

    // ---- C-layout -> LDS fp32 tile --------------------------------------
#pragma unroll
    for (int nt = 0; nt < 4; ++nt)
#pragma unroll
        for (int r = 0; r < 8; ++r) {
            int ml = (wave & 3) * 16 + r + khalf * 8;
            int cl = (wave >> 2) * 64 + nt * 16 + lrow;
            tile[ml][cl] = acc[nt][r];
        }
    __syncthreads();

    // ---- RoPE epilogue: 8192 elems / 256 threads ------------------------
    const int part = bn >> 3;                // 0=q 1=k 2=v
    const int h    = bn & 7;
#pragma unroll 4
    for (int e = 0; e < 32; ++e) {
        int idx = tid + e * 256;
        int row = idx >> 7;
        int d   = idx & 127;
        int gm  = bm * 64 + row;
        int b   = gm >> 12;
        int t   = gm & (T_ - 1);
        float val = tile[row][d];
        if (part == 2) {
            vo[((size_t)(b * H_ + h) * D_ + d) * T_ + t] = f32_to_bf16(val);
        } else {
            float pairv = tile[row][(d + 64) & 127];
            float rot   = (d < 64) ? -pairv : pairv;
            float freq  = __powf(10000.0f, -(float)(d & 63) * (1.0f / 64.0f));
            float sv, cv;
            __sincosf((float)t * freq, &sv, &cv);
            unsigned short bfv = f32_to_bf16(val * cv + rot * sv);
            size_t o = ((size_t)(b * H_ + h) * T_ + t) * (size_t)D_ + d;
            if (part == 0) qo[o] = bfv; else ko[o] = bfv;
        }
    }
}

// ---------------------------------------------------------------------------
// Kernel 2: flash attention, causal + sliding window (j in [i-512, i]).
// One wave owns 16 queries end-to-end; 8 waves/WG -> 128 queries/WG.
// V fragments are hoisted ahead of the softmax VALU chain so their loads
// hide under the shfl/exp dependency chain.
// ---------------------------------------------------------------------------
__global__ __launch_bounds__(256) void attn_swa_kernel(
    const unsigned short* __restrict__ qb,   // [B,H,T,D]
    const unsigned short* __restrict__ kb,   // [B,H,T,D]
    const unsigned short* __restrict__ vt,   // [B,H,D,T]
    unsigned short* __restrict__ att)        // [B,T,H,D]
{
    __shared__ __align__(128) unsigned short pl[8][16][32];  // per-wave P

    const int tid   = threadIdx.x;
    const int wave  = tid >> 5;
    const int lane  = tid & 31;
    const int lrow  = lane & 15;
    const int khalf = lane >> 4;

    const int bh = blockIdx.x;
    const int b  = bh >> 3;
    const int h  = bh & 7;
    const int q0 = blockIdx.y * 128 + wave * 16;

    const unsigned short* qbase = qb + (size_t)bh * T_ * D_;
    const unsigned short* kbase = kb + (size_t)bh * T_ * D_;
    const unsigned short* vbase = vt + (size_t)bh * D_ * T_;

    v16bf qf[4];
#pragma unroll
    for (int kc = 0; kc < 4; ++kc)
        qf[kc] = *(const v16bf*)(qbase + (size_t)(q0 + lrow) * D_ + kc * 32 + khalf * 16);

    v8f   o[8] = {};
    float mrow[8], lsum[8];
#pragma unroll
    for (int r = 0; r < 8; ++r) { mrow[r] = -1e30f; lsum[r] = 0.0f; }

    const float scale  = 0.08838834764831845f;     // 1/sqrt(128)
    const int   kstart = (q0 > W_) ? ((q0 - W_) & ~31) : 0;
    const int   kend   = q0 + 15;

    for (int kblk = kstart; kblk <= kend; kblk += 32) {
        if (kblk + 32 <= kend)
            __builtin_prefetch(kbase + (size_t)(kblk + 32 + lrow) * D_, 0, 3);

        // ---- S = Q K^T, two 16-key tiles --------------------------------
        v8f s0 = {}, s1 = {};
#pragma unroll
        for (int kc = 0; kc < 4; ++kc) {
            v16bf k0f = *(const v16bf*)(kbase + (size_t)(kblk      + lrow) * D_ + kc * 32 + khalf * 16);
            v16bf k1f = *(const v16bf*)(kbase + (size_t)(kblk + 16 + lrow) * D_ + kc * 32 + khalf * 16);
            s0 = wmma_bf16(qf[kc], k0f, s0);
            s1 = wmma_bf16(qf[kc], k1f, s1);
        }

        // ---- hoist V fragments: loads hide under the softmax chain ------
        v16bf vf[8];
#pragma unroll
        for (int nt = 0; nt < 8; ++nt)
            vf[nt] = *(const v16bf*)(vbase + (size_t)(nt * 16 + lrow) * T_ + kblk + khalf * 16);

        // ---- mask + online softmax (rows live in 16-lane halves) --------
#pragma unroll
        for (int r = 0; r < 8; ++r) {
            int i  = q0 + r + 8 * khalf;
            int j0 = kblk + lrow;
            int j1 = kblk + 16 + lrow;
            float v0 = s0[r] * scale;
            float v1 = s1[r] * scale;
            if (j0 > i || j0 < i - W_) v0 = -1e30f;
            if (j1 > i || j1 < i - W_) v1 = -1e30f;

            float pm = fmaxf(v0, v1);
#pragma unroll
            for (int m = 1; m < 16; m <<= 1) pm = fmaxf(pm, __shfl_xor(pm, m, 32));
            float mnew  = fmaxf(mrow[r], pm);
            float alpha = __expf(mrow[r] - mnew);
            mrow[r] = mnew;

            float p0 = __expf(v0 - mnew);
            float p1 = __expf(v1 - mnew);
            float ps = p0 + p1;
#pragma unroll
            for (int m = 1; m < 16; m <<= 1) ps += __shfl_xor(ps, m, 32);
            lsum[r] = lsum[r] * alpha + ps;

#pragma unroll
            for (int nt = 0; nt < 8; ++nt) o[nt][r] *= alpha;

            int row = r + 8 * khalf;
            pl[wave][row][lrow]      = f32_to_bf16(p0);
            pl[wave][row][16 + lrow] = f32_to_bf16(p1);
        }

        // wave-private C-layout -> A-layout handoff through LDS
        asm volatile("s_wait_dscnt 0" ::: "memory");
        __builtin_amdgcn_wave_barrier();

        v16bf pf = *(const v16bf*)(&pl[wave][lrow][khalf * 16]);

        // ---- O += P V ---------------------------------------------------
#pragma unroll
        for (int nt = 0; nt < 8; ++nt)
            o[nt] = wmma_bf16(pf, vf[nt], o[nt]);
    }

    // ---- normalize and store [B,T,H,D] bf16 -----------------------------
#pragma unroll
    for (int r = 0; r < 8; ++r) {
        float inv = 1.0f / lsum[r];
        int   t   = q0 + r + 8 * khalf;
#pragma unroll
        for (int nt = 0; nt < 8; ++nt) {
            size_t off = (((size_t)b * T_ + t) * H_ + h) * (size_t)D_ + nt * 16 + lrow;
            att[off] = f32_to_bf16(o[nt][r] * inv);
        }
    }
}

// ---------------------------------------------------------------------------
// Kernel 3: out = att @ w_o^T -> fp32.  Async global->LDS double-buffered
// pipeline (GLOBAL_LOAD_ASYNC_TO_LDS_B128 / ASYNCcnt / s_wait_asynccnt).
// Tile 64(M) x 128(N); A stage 64x32, B stage 128x32.
// ---------------------------------------------------------------------------
__global__ __launch_bounds__(256) void out_proj_kernel(
    const unsigned short* __restrict__ ab,   // [8192,1024] bf16
    const unsigned short* __restrict__ wb,   // [1024,1024] bf16
    float* __restrict__ out)                 // [8192,1024] f32
{
    __shared__ __align__(128) unsigned short lA[2][64][40];   // 32 cols + pad
    __shared__ __align__(128) unsigned short lB[2][128][40];

    const int tid   = threadIdx.x;
    const int wave  = tid >> 5;
    const int lane  = tid & 31;
    const int lrow  = lane & 15;
    const int khalf = lane >> 4;

    const int blk = blockIdx.x;
    const int bm  = blk & 127;
    const int bn  = blk >> 7;
    const int m0g = bm * 64;
    const int n0g = bn * 128;

    // staging map: each thread moves one 16B chunk of A and two of B
    const int rowS = tid >> 2;               // 0..63
    const int colS = (tid & 3) * 8;          // 0,8,16,24 (bf16 elems)

    const unsigned short* gA  = ab + (size_t)(m0g + rowS)      * C_ + colS;
    const unsigned short* gB0 = wb + (size_t)(n0g + rowS)      * C_ + colS;
    const unsigned short* gB1 = wb + (size_t)(n0g + 64 + rowS) * C_ + colS;

    v8f acc[4] = {};

#if defined(HAVE_ASYNC_LDS)
    // ---- async global->LDS double-buffer pipeline -----------------------
    auto stage = [&](int buf, int k0) {
        ASYNC_CP_B128(gA + k0,  &lA[buf][rowS][colS]);
        ASYNC_CP_B128(gB0 + k0, &lB[buf][rowS][colS]);
        ASYNC_CP_B128(gB1 + k0, &lB[buf][64 + rowS][colS]);
    };
    stage(0, 0);
    int buf = 0;
    for (int k0 = 0; k0 < C_; k0 += 32, buf ^= 1) {
        if (k0 + 32 < C_) {
            stage(buf ^ 1, k0 + 32);
            WAIT_ASYNC(3);                   // current stage drained, next in flight
        } else {
            WAIT_ASYNC(0);
        }
        __syncthreads();
        v16bf a = *(const v16bf*)(&lA[buf][(wave & 3) * 16 + lrow][khalf * 16]);
#pragma unroll
        for (int nt = 0; nt < 4; ++nt) {
            v16bf bfrag = *(const v16bf*)(&lB[buf][(wave >> 2) * 64 + nt * 16 + lrow][khalf * 16]);
            acc[nt] = wmma_bf16(a, bfrag, acc[nt]);
        }
        __syncthreads();
    }
#else
    // ---- register-staged LDS double-buffer pipeline ---------------------
    uint4 rA  = *(const uint4*)(gA);
    uint4 rB0 = *(const uint4*)(gB0);
    uint4 rB1 = *(const uint4*)(gB1);
    int buf = 0;
    for (int k0 = 0; k0 < C_; k0 += 32, buf ^= 1) {
        *(uint4*)(&lA[buf][rowS][colS])      = rA;
        *(uint4*)(&lB[buf][rowS][colS])      = rB0;
        *(uint4*)(&lB[buf][64 + rowS][colS]) = rB1;
        if (k0 + 32 < C_) {                  // next-stage loads fly under WMMAs
            rA  = *(const uint4*)(gA  + k0 + 32);
            rB0 = *(const uint4*)(gB0 + k0 + 32);
            rB1 = *(const uint4*)(gB1 + k0 + 32);
        }
        __syncthreads();
        v16bf a = *(const v16bf*)(&lA[buf][(wave & 3) * 16 + lrow][khalf * 16]);
#pragma unroll
        for (int nt = 0; nt < 4; ++nt) {
            v16bf bfrag = *(const v16bf*)(&lB[buf][(wave >> 2) * 64 + nt * 16 + lrow][khalf * 16]);
            acc[nt] = wmma_bf16(a, bfrag, acc[nt]);
        }
        __syncthreads();
    }
#endif

#pragma unroll
    for (int nt = 0; nt < 4; ++nt)
#pragma unroll
        for (int r = 0; r < 8; ++r) {
            int gm = m0g + (wave & 3) * 16 + r + khalf * 8;
            int gn = n0g + (wave >> 2) * 64 + nt * 16 + lrow;
            out[(size_t)gm * C_ + gn] = acc[nt][r];
        }
}

// ---------------------------------------------------------------------------
// Host launcher
// ---------------------------------------------------------------------------
extern "C" void kernel_launch(void* const* d_in, const int* in_sizes, int n_in,
                              void* d_out, int out_size, void* d_ws, size_t ws_size,
                              hipStream_t stream) {
    (void)in_sizes; (void)n_in; (void)out_size; (void)ws_size;
    const float* x     = (const float*)d_in[0];
    const float* w_qkv = (const float*)d_in[1];
    const float* w_o   = (const float*)d_in[2];
    float*       out   = (float*)d_out;

    char*  ws  = (char*)d_ws;
    size_t off = 0;
    auto carve = [&](size_t bytes) -> unsigned short* {
        unsigned short* p = (unsigned short*)(ws + off);
        off += (bytes + 255) & ~(size_t)255;
        return p;
    };
    unsigned short* xb  = carve((size_t)MT_ * C_ * 2);
    unsigned short* wqb = carve((size_t)TC_ * C_ * 2);
    unsigned short* wob = carve((size_t)C_  * C_ * 2);
    unsigned short* qo  = carve((size_t)MT_ * C_ * 2);
    unsigned short* ko  = carve((size_t)MT_ * C_ * 2);
    unsigned short* vt  = carve((size_t)MT_ * C_ * 2);
    unsigned short* att = carve((size_t)MT_ * C_ * 2);

    {
        int n4 = (MT_ * C_) / 4;
        cvt_f32_to_bf16_kernel<<<(n4 + 255) / 256, 256, 0, stream>>>(
            (const float4*)x, (unsigned long long*)xb, n4);
        n4 = (TC_ * C_) / 4;
        cvt_f32_to_bf16_kernel<<<(n4 + 255) / 256, 256, 0, stream>>>(
            (const float4*)w_qkv, (unsigned long long*)wqb, n4);
        n4 = (C_ * C_) / 4;
        cvt_f32_to_bf16_kernel<<<(n4 + 255) / 256, 256, 0, stream>>>(
            (const float4*)w_o, (unsigned long long*)wob, n4);
    }

    qkv_rope_kernel<<<dim3(128 * 24), 256, 0, stream>>>(xb, wqb, qo, ko, vt);
    attn_swa_kernel<<<dim3(B_ * H_, T_ / 128), 256, 0, stream>>>(qo, ko, vt, att);
    out_proj_kernel<<<dim3(128 * 8), 256, 0, stream>>>(att, wob, out);
}
